// AFTFull_49649821942237
// MI455X (gfx1250) — matrix operand order repaired and checked by reading, not compiled, not run on hardware
//
#include <hip/hip_runtime.h>

// ---------------------------------------------------------------------------
// AFT-Full forward for MI455X (gfx1250), bf16 WMMA pipeline, fragment-order
// layouts (32B contiguous per lane), TDM double-buffered staging in phase B.
// B=8, DIM=512, T=4096, H=64.
// ---------------------------------------------------------------------------

typedef __bf16 bf16;
typedef __attribute__((ext_vector_type(16))) __bf16 v16bf;
typedef __attribute__((ext_vector_type(8)))  float  v8f;
typedef __attribute__((ext_vector_type(4)))  unsigned int u32x4;
typedef __attribute__((ext_vector_type(8)))  int  i32x8;
typedef __attribute__((ext_vector_type(4)))  int  i32x4;

#define WMMA_BF16(a, b, c) \
  __builtin_amdgcn_wmma_f32_16x16x32_bf16(false, (a), false, (b), (short)0, (c), false, false)

// --- fragment-order index helpers (CDNA5 ISA 7.12.2, wave32) ---------------
// A (16x32 bf16): lane = (m&15) | (k&8 ? 16 : 0); e = (k&7) | ((k&16)>>1)
__device__ __forceinline__ int a_frag_idx(int m, int k) {
  const int sub  = m >> 4;
  const int lane = (m & 15) | ((k & 8) << 1);
  const int e    = (k & 7) | ((k & 16) >> 1);
  return ((sub * 32 + lane) << 4) | e;
}
// B (32x16 bf16): lane = (n&15) | (k&8 ? 16 : 0); e = (k&7) | ((k&16)>>1)
__device__ __forceinline__ int b_frag_idx(int k, int n) {
  const int j    = n >> 4;
  const int lane = (n & 15) | ((k & 8) << 1);
  const int e    = (k & 7) | ((k & 16) >> 1);
  return ((j * 32 + lane) << 4) | e;
}
// C/D f32 16x16: lanes 0-15: N=lane, reg r: M=r ; lanes 16-31: N=lane-16, M=r+8

// TDM: DMA a contiguous `n8*8`-byte block from global into LDS.
// D# per CDNA5 ISA ch.8 (group0 128b, group1 256b; groups 2/3 unused -> 0).
// This toolchain's builtin is the 6-arg form:
//   (u32x4 g0, i32x8 g1, i32x4 g2, i32x4 g3, i32x8 extra, i32 cpol)
__device__ __forceinline__ void tdm_load_tile(unsigned lds_off,
                                              const void* gptr,
                                              unsigned n8 /*8B units*/) {
  const unsigned long long ga = (unsigned long long)(uintptr_t)gptr;
  u32x4 g0;
  g0[0] = 1u;                                            // count=1 (valid D#)
  g0[1] = lds_off;                                       // lds_addr (bytes)
  g0[2] = (unsigned)(ga & 0xFFFFFFFFu);                  // global_addr[31:0]
  g0[3] = (unsigned)((ga >> 32) & 0x01FFFFFFu) | (2u << 30);  // addr hi|type=2
  i32x8 g1;
  g1[0] = (int)(3u << 16);          // data_size=3 (8B), no multicast/pad/iter
  g1[1] = 0;                        // atomic_barrier_addr=0, tensor_dim0 lo16=0
  g1[2] = (int)(0x4000u | (1u << 16)); // tensor_dim0 hi16 (2^30), dim1 lo16=1
  g1[3] = (int)(n8 << 16);          // dim1 hi16=0, tile_dim0 = n8
  g1[4] = 0;                        // tile_dim1=0 (unused), tile_dim2=0
  g1[5] = (int)n8;                  // tensor_dim0_stride lo32 (unused for 1D)
  g1[6] = 0;
  g1[7] = 0;
  const i32x4 z4 = {0, 0, 0, 0};
  const i32x8 z8 = {0, 0, 0, 0, 0, 0, 0, 0};
  __builtin_amdgcn_tensor_load_to_lds(g0, g1, z4, z4, z8, 0);
}

// ===========================================================================
// Phase A: Q,K,V projections.  Block: 128 t-rows x 192 outputs (q|k|v).
// Emits sigQ (f32, linear) and R = [eK*V | eK] (bf16, B-fragment order per
// (b, s-chunk of 32): R[(b*128+c)*4096 + b_frag_idx(s&31, n)]).
// ===========================================================================
__global__ __launch_bounds__(256) void qkv_kernel(
    const float* __restrict__ x,
    const float* __restrict__ wq, const float* __restrict__ bq,
    const float* __restrict__ wk, const float* __restrict__ bk,
    const float* __restrict__ wv, const float* __restrict__ bv,
    float* __restrict__ sigQ, bf16* __restrict__ R) {
  __shared__ __align__(16) bf16 sXf[8 * 32 * 16];    // A frags: 8 subtiles
  __shared__ __align__(16) bf16 sWf[12 * 32 * 16];   // B frags: 12 subtiles
  const int tid  = threadIdx.x;
  const int wave = tid >> 5;
  const int lane = tid & 31;
  const int b    = blockIdx.x >> 5;           // 8 batches
  const int t0   = (blockIdx.x & 31) * 128;   // 32 m-tiles

  v8f acc[12];
#pragma unroll
  for (int j = 0; j < 12; ++j)
#pragma unroll
    for (int r = 0; r < 8; ++r) acc[j][r] = 0.0f;

  for (int kc = 0; kc < 16; ++kc) {           // DIM=512 in chunks of 32
    const int k0 = kc * 32;
    __syncthreads();
    // stage x tile (coalesced float4 reads, swizzled bf16 writes)
    for (int u = tid; u < 32 * 32; u += 256) {
      const int kk = u >> 5, t4 = (u & 31) * 4;
      const float4 v =
          *(const float4*)&x[((size_t)b * 512 + (k0 + kk)) * 4096 + (t0 + t4)];
      const float* vp = (const float*)&v;
#pragma unroll
      for (int i = 0; i < 4; ++i)
        sXf[a_frag_idx(t4 + i, kk)] = (bf16)vp[i];
    }
    // stage stacked weights in B-fragment order: one (j,lane) per unit
    for (int u = tid; u < 12 * 32; u += 256) {
      const int j = u >> 5, ln = u & 31;
      const int n  = j * 16 + (ln & 15);
      const int kr = k0 + ((ln & 16) ? 8 : 0);
      const float* wrow = (n < 64)  ? (wq + (size_t)n * 512)
                        : (n < 128) ? (wk + (size_t)(n - 64) * 512)
                                    : (wv + (size_t)(n - 128) * 512);
      v16bf tmp;
#pragma unroll
      for (int e = 0; e < 16; ++e) tmp[e] = (bf16)wrow[kr + (e & 7) + ((e & 8) << 1)];
      *(v16bf*)(sWf + (u << 4)) = tmp;
    }
    __syncthreads();
    const v16bf a = *(const v16bf*)(sXf + ((wave * 32 + lane) << 4));
#pragma unroll
    for (int j = 0; j < 12; ++j) {
      const v16bf bb = *(const v16bf*)(sWf + ((j * 32 + lane) << 4));
      acc[j] = WMMA_BF16(a, bb, acc[j]);
    }
  }

  // epilogue: sigmoid(Q); pack exp(K)*V and exp(K) into fragment-order R
  const int nl  = lane & 15;
  const int mhi = (lane < 16) ? 0 : 8;
#pragma unroll
  for (int r = 0; r < 8; ++r) {
    const int    t    = t0 + wave * 16 + r + mhi;
    const size_t row  = (size_t)b * 4096 + t;
    const size_t base = (size_t)(b * 128 + (t >> 5)) << 12;  // *4096
    const int    ks   = t & 31;
#pragma unroll
    for (int j = 0; j < 4; ++j) {
      const int h = j * 16 + nl;
      const float q = acc[j][r] + bq[h];
      sigQ[row * 64 + h] = 1.0f / (1.0f + __expf(-q));
      const float kk = acc[4 + j][r] + bk[h];
      const float vv = acc[8 + j][r] + bv[h];
      const float ek = __expf(kk);
      R[base + b_frag_idx(ks, h)]      = (bf16)(ek * vv);
      R[base + b_frag_idx(ks, 64 + h)] = (bf16)ek;
    }
  }
}

// ===========================================================================
// Phase B: [num|den] = exp(wbias[t,:]) @ R[b,:,0:128]; Yt = sigQ*num/den.
// A fragments built in registers straight from wbias (2x 8-float runs/lane).
// B tiles (8KB, fragment order) DMA'd to LDS by the Tensor Data Mover,
// double-buffered: wave0 issues TENSOR_LOAD_TO_LDS for chunk kc+1 while all
// waves run WMMAs on chunk kc; s_wait_tensorcnt + barrier at iteration end.
// Yt written in A-fragment order for phase C.
// ===========================================================================
__global__ __launch_bounds__(256) void aft_kernel(
    const float* __restrict__ wbias, const bf16* __restrict__ R,
    const float* __restrict__ sigQ, bf16* __restrict__ Yt) {
  __shared__ __align__(16) bf16 sR[2][8 * 32 * 16];   // 2 x 8KB tiles
  const int tid  = threadIdx.x;
  const int wave = tid >> 5;
  const int lane = tid & 31;
  const int b    = blockIdx.x >> 5;
  const int t0   = (blockIdx.x & 31) * 128;
  const unsigned lds0 = (unsigned)(uintptr_t)(void*)&sR[0][0];

  v8f acc[8];
#pragma unroll
  for (int j = 0; j < 8; ++j)
#pragma unroll
    for (int r = 0; r < 8; ++r) acc[j][r] = 0.0f;

  // per-lane wbias row pointer for A fragments
  const float* wrow0 =
      wbias + (size_t)(t0 + wave * 16 + (lane & 15)) * 4096 +
      ((lane & 16) ? 8 : 0);

  // prologue: DMA tile 0 into buffer 0
  if (wave == 0) {
    tdm_load_tile(lds0, R + ((size_t)(b * 128) << 12), 1024);
    __builtin_amdgcn_s_wait_tensorcnt(0);
  }
  __syncthreads();

  for (int kc = 0; kc < 128; ++kc) {
    const int cur = kc & 1;
    // kick DMA for next tile into the other buffer
    if (wave == 0 && kc + 1 < 128)
      tdm_load_tile(lds0 + (unsigned)((cur ^ 1) * 8192),
                    R + ((size_t)(b * 128 + kc + 1) << 12), 1024);

    // A fragment in registers: exp of two 8-float runs of one wbias row
    const float* wrow = wrow0 + kc * 32;
    const float4 r0 = *(const float4*)(wrow);
    const float4 r1 = *(const float4*)(wrow + 4);
    const float4 r2 = *(const float4*)(wrow + 16);
    const float4 r3 = *(const float4*)(wrow + 20);
    const float* f0 = (const float*)&r0;
    const float* f1 = (const float*)&r1;
    const float* f2 = (const float*)&r2;
    const float* f3 = (const float*)&r3;
    v16bf a;
#pragma unroll
    for (int i = 0; i < 4; ++i) {
      a[i]      = (bf16)__expf(f0[i]);
      a[4 + i]  = (bf16)__expf(f1[i]);
      a[8 + i]  = (bf16)__expf(f2[i]);
      a[12 + i] = (bf16)__expf(f3[i]);
    }

    // preload all 8 B fragments, then run the WMMAs
    v16bf bbv[8];
#pragma unroll
    for (int j = 0; j < 8; ++j)
      bbv[j] = *(const v16bf*)(&sR[cur][0] + ((j * 32 + lane) << 4));
#pragma unroll
    for (int j = 0; j < 8; ++j) acc[j] = WMMA_BF16(a, bbv[j], acc[j]);

    // next tile ready before anyone reads it
    if (wave == 0 && kc + 1 < 128) __builtin_amdgcn_s_wait_tensorcnt(0);
    __syncthreads();
  }

  // epilogue: Yt = sigmoid(Q) * num / den, stored in A-fragment order
  const int nl  = lane & 15;
  const int mhi = (lane < 16) ? 0 : 8;
  const int m16 = ((b * 4096 + t0) >> 4) + wave;
#pragma unroll
  for (int r = 0; r < 8; ++r) {
    const int    mloc = r + mhi;
    const size_t row  = (size_t)b * 4096 + t0 + wave * 16 + mloc;
#pragma unroll
    for (int j = 0; j < 4; ++j) {
      const int   h = j * 16 + nl;
      const float y = sigQ[row * 64 + h] * acc[j][r] / acc[4 + j][r];
      const int   c      = j >> 1;                     // h>>5
      const int   lane_a = mloc | ((nl & 8) << 1);
      const int   e      = (nl & 7) | ((j & 1) << 3);
      Yt[(((size_t)(m16 * 2 + c) * 32 + lane_a) << 4) | e] = (bf16)y;
    }
  }
}

// ===========================================================================
// Phase C: out = Yt @ wp.T + bp, written twice (reference returns (out,out)).
// A fragments loaded directly from global (already fragment order).
// ===========================================================================
__global__ __launch_bounds__(256) void proj_kernel(
    const bf16* __restrict__ Yt, const float* __restrict__ wp,
    const float* __restrict__ bp, float* __restrict__ out0,
    float* __restrict__ out1) {
  __shared__ __align__(16) bf16 sB[8 * 32 * 16];   // 8KB: one 32x128 wp tile
  const int tid  = threadIdx.x;
  const int wave = tid >> 5;
  const int lane = tid & 31;
  const int bt0  = (blockIdx.x >> 2) * 128;   // 256 m-blocks over B*T=32768
  const int n0b  = (blockIdx.x & 3) * 128;    // 4 n-blocks over DIM=512

  v8f acc[8];
#pragma unroll
  for (int j = 0; j < 8; ++j)
#pragma unroll
    for (int r = 0; r < 8; ++r) acc[j][r] = 0.0f;

#pragma unroll
  for (int kc = 0; kc < 2; ++kc) {
    __syncthreads();
    // stage wp tile in B-fragment order: (j,lane) = tid, two 8-f32 runs 16 apart
    {
      const int n  = (tid >> 5) * 16 + (tid & 15);
      const int kr = kc * 32 + ((tid & 16) ? 8 : 0);
      const float* wrow = wp + (size_t)(n0b + n) * 64 + kr;
      v16bf tmp;
#pragma unroll
      for (int e = 0; e < 16; ++e) tmp[e] = (bf16)wrow[(e & 7) + ((e & 8) << 1)];
      *(v16bf*)(sB + (tid << 4)) = tmp;
    }
    __syncthreads();
    const v16bf a = *(const v16bf*)(
        Yt + (((size_t)(((bt0 >> 4) + wave) * 2 + kc) * 32 + lane) << 4));
#pragma unroll
    for (int j = 0; j < 8; ++j) {
      const v16bf bb = *(const v16bf*)(sB + ((j * 32 + lane) << 4));
      acc[j] = WMMA_BF16(a, bb, acc[j]);
    }
  }

  const int nl  = lane & 15;
  const int mhi = (lane < 16) ? 0 : 8;
#pragma unroll
  for (int r = 0; r < 8; ++r) {
    const int bt = bt0 + wave * 16 + r + mhi;
#pragma unroll
    for (int j = 0; j < 8; ++j) {
      const int   d   = n0b + j * 16 + nl;
      const float val = acc[j][r] + bp[d];
      out0[(size_t)bt * 512 + d] = val;
      out1[(size_t)bt * 512 + d] = val;
    }
  }
}

// ===========================================================================
extern "C" void kernel_launch(void* const* d_in, const int* in_sizes, int n_in,
                              void* d_out, int out_size, void* d_ws,
                              size_t ws_size, hipStream_t stream) {
  const float* x     = (const float*)d_in[0];
  const float* wq    = (const float*)d_in[1];
  const float* bq    = (const float*)d_in[2];
  const float* wk    = (const float*)d_in[3];
  const float* bk    = (const float*)d_in[4];
  const float* wv    = (const float*)d_in[5];
  const float* bv    = (const float*)d_in[6];
  const float* wp    = (const float*)d_in[7];
  const float* bp    = (const float*)d_in[8];
  const float* wbias = (const float*)d_in[9];

  float* out = (float*)d_out;

  // workspace layout
  char*  ws   = (char*)d_ws;
  bf16*  R    = (bf16*)ws;                                   // 8 MB (frag order)
  float* sigQ = (float*)(ws + (size_t)8 * 4096 * 128 * 2);   // 8 MB linear
  bf16*  Yt   = (bf16*)(ws + (size_t)8 * 4096 * 128 * 2 +
                        (size_t)8 * 4096 * 64 * 4);          // 4 MB (frag order)

  qkv_kernel<<<256, 256, 0, stream>>>(x, wq, bq, wk, bk, wv, bv, sigQ, R);
  aft_kernel<<<256, 256, 0, stream>>>(wbias, R, sigQ, Yt);
  proj_kernel<<<1024, 256, 0, stream>>>(Yt, wp, bp, out,
                                        out + (size_t)8 * 4096 * 512);
}
